// CompleteVQVAETextModel_34840774705495
// MI455X (gfx1250) — compile-verified, matrix operand order checked by reading, not run on hardware
//
#include <hip/hip_runtime.h>
#include <hip/hip_bf16.h>

typedef __attribute__((ext_vector_type(16))) __bf16 v16bf;
typedef __attribute__((ext_vector_type(8)))  __bf16 v8bf;
typedef __attribute__((ext_vector_type(8)))  float  v8f;

#define BN_EPS 1e-5f
#define LRELU_SLOPE 0.01f

// Problem dims (hard-coded per reference)
#define BB     32
#define L_IN   4096
#define C_IN0  256
#define L1D    2048
#define C1D    256
#define L2D    1024
#define C2D    512
#define L3D    1024
#define C3D    128
#define KCODES 4096
#define DDIM   128
#define NROWS  (BB * L3D)   // 32768 z-rows

// ---------------------------------------------------------------------------
// Tile loader: per-lane fragment for WMMA bf16 16x16x32.
// A (16x32, row-major, K contiguous): lane = row (lane&15), lane-half picks
// K offset 0/8, VGPRs 4-7 hold K+16..K+23 (ISA 7.12.2). B stored as B^T
// (N-major, K contiguous) uses the identical pattern with lane = column.
// Works for both global and LDS source pointers (addrspace inferred).
// ---------------------------------------------------------------------------
__device__ __forceinline__ v16bf load_frag(const __bf16* rowptr, int kbase, int lane) {
  const int kk = (lane >> 4) << 3;         // 0 or 8
  v8bf lo = *(const v8bf*)(rowptr + kbase + kk);
  v8bf hi = *(const v8bf*)(rowptr + kbase + kk + 16);
  return __builtin_shufflevector(lo, hi, 0,1,2,3,4,5,6,7,8,9,10,11,12,13,14,15);
}

__device__ __forceinline__ v16bf zero_frag() {
  v16bf z;
#pragma unroll
  for (int i = 0; i < 16; ++i) z[i] = (__bf16)0.0f;
  return z;
}

// ---------------------------------------------------------------------------
// CDNA5 async memory->LDS copy (tracked by ASYNCcnt). 16 bytes per lane.
// Inline asm keeps this portable across ROCm7.2 / amdgpu-toolchain builtins.
// ---------------------------------------------------------------------------
__device__ __forceinline__ void async_copy_b128(const void* gsrc, void* ldst) {
  unsigned ldsoff = (unsigned)(unsigned long long)ldst;   // flat LDS addr[31:0] = offset
  unsigned long long ga = (unsigned long long)gsrc;
  asm volatile("global_load_async_to_lds_b128 %0, %1, off"
               :: "v"(ldsoff), "v"(ga) : "memory");
}

__device__ __forceinline__ void wait_async0() {
  asm volatile("s_wait_asynccnt 0x0" ::: "memory");
}

// ---------------------------------------------------------------------------
// Prep: transpose x [B][C][L] f32 -> channel-last bf16 [B][L][C]
// ---------------------------------------------------------------------------
__global__ void k_transpose_x(const float* __restrict__ x, __bf16* __restrict__ xcl) {
  size_t idx = (size_t)blockIdx.x * blockDim.x + threadIdx.x;
  const size_t total = (size_t)BB * C_IN0 * L_IN;
  if (idx >= total) return;
  int t = (int)(idx % L_IN);
  int c = (int)((idx / L_IN) % C_IN0);
  int b = (int)(idx / ((size_t)L_IN * C_IN0));
  xcl[((size_t)b * L_IN + t) * C_IN0 + c] = (__bf16)x[idx];
}

// ---------------------------------------------------------------------------
// Prep: fold BN scale into conv weights, repack [O][I][3] -> [k][O][I] bf16,
// fold BN shift into bias (f32).
// ---------------------------------------------------------------------------
__global__ void k_prep_w(const float* __restrict__ w, const float* __restrict__ bin,
                         const float* __restrict__ g, const float* __restrict__ beta,
                         const float* __restrict__ mean, const float* __restrict__ var,
                         __bf16* __restrict__ wk, float* __restrict__ bout,
                         int O, int I, int has_bn) {
  int idx = blockIdx.x * blockDim.x + threadIdx.x;
  int total = O * I * 3;
  if (idx < total) {
    int k = idx % 3;
    int i = (idx / 3) % I;
    int o = idx / (3 * I);
    float s = has_bn ? g[o] * rsqrtf(var[o] + BN_EPS) : 1.0f;
    wk[((size_t)k * O + o) * I + i] = (__bf16)(w[idx] * s);
  }
  if (idx < O) {
    if (has_bn) {
      float s = g[idx] * rsqrtf(var[idx] + BN_EPS);
      bout[idx] = (bin[idx] - mean[idx]) * s + beta[idx];
    } else {
      bout[idx] = bin[idx];
    }
  }
}

// ---------------------------------------------------------------------------
// Prep: codebook f32 -> bf16 + per-row ||e||^2 (from bf16 values, consistent)
// ---------------------------------------------------------------------------
__global__ void k_prep_cb(const float* __restrict__ cb, __bf16* __restrict__ cbb,
                          float* __restrict__ esq) {
  int r = blockIdx.x * blockDim.x + threadIdx.x;
  if (r >= KCODES) return;
  float s = 0.0f;
#pragma unroll 8
  for (int d = 0; d < DDIM; ++d) {
    __bf16 bv = (__bf16)cb[(size_t)r * DDIM + d];
    cbb[(size_t)r * DDIM + d] = bv;
    float f = (float)bv;
    s += f * f;
  }
  esq[r] = s;
}

// ---------------------------------------------------------------------------
// ||z||^2 per row
// ---------------------------------------------------------------------------
__global__ void k_row_sq(const __bf16* __restrict__ z, float* __restrict__ zsq) {
  int r = blockIdx.x * blockDim.x + threadIdx.x;
  if (r >= NROWS) return;
  float s = 0.0f;
#pragma unroll 8
  for (int d = 0; d < DDIM; ++d) {
    float f = (float)z[(size_t)r * DDIM + d];
    s += f * f;
  }
  zsq[r] = s;
}

// ---------------------------------------------------------------------------
// Implicit-GEMM conv1d (pad=1, kernel=3) + optional LeakyReLU, bf16 in/out.
// Activations channel-last [B][L][C]; weights [k][COUT][CIN] (BN pre-folded).
// One wave computes a 16(out-ch) x 16(time) output tile:
//   conv = sum over 3 taps of GEMM(COUT x CIN x 16), K stepped 32 per WMMA.
// ---------------------------------------------------------------------------
template<int CIN, int COUT, int LIN, int LOUT, int STRIDE, bool DO_LRELU>
__global__ void k_conv_wmma(const __bf16* __restrict__ in,
                            const __bf16* __restrict__ wk,
                            const float* __restrict__ bias,
                            __bf16* __restrict__ out) {
  const int lane = threadIdx.x & 31;
  const int wave = blockIdx.x * (blockDim.x >> 5) + (threadIdx.x >> 5);
  const int OT = COUT / 16, TT = LOUT / 16;
  if (wave >= BB * TT * OT) return;
  const int ot = wave % OT;
  const int tt = (wave / OT) % TT;
  const int b  = wave / (OT * TT);
  const int obase = ot * 16, tbase = tt * 16;
  const int n   = lane & 15;          // A row (out-ch) and B col (time) index
  const int hi8 = (lane >> 4) << 3;   // C/D: this lane holds rows hi8..hi8+7

  v8f acc;
#pragma unroll
  for (int j = 0; j < 8; ++j) acc[j] = 0.0f;

  const int t = tbase + n;
#pragma unroll
  for (int k = 0; k < 3; ++k) {
    const int tin = STRIDE * t + k - 1;
    const bool valid = (tin >= 0) && (tin < LIN);
    const __bf16* aK = wk + ((size_t)k * COUT + obase + n) * CIN;
    const __bf16* bK = in + ((size_t)b * LIN + (valid ? tin : 0)) * CIN;
#pragma unroll
    for (int kc = 0; kc < CIN; kc += 32) {
      v16bf a = load_frag(aK, kc, lane);
      v16bf bt = valid ? load_frag(bK, kc, lane) : zero_frag();
      acc = __builtin_amdgcn_wmma_f32_16x16x32_bf16(
          false, a, false, bt, (short)0, acc, false, false);
    }
  }

  // Epilogue: bias + LeakyReLU, 8 contiguous out-channels -> one 16B store.
  const int o0 = obase + hi8;
  v8bf res;
#pragma unroll
  for (int j = 0; j < 8; ++j) {
    float v = acc[j] + bias[o0 + j];
    if (DO_LRELU) v = (v >= 0.0f) ? v : LRELU_SLOPE * v;
    res[j] = (__bf16)v;
  }
  *(v8bf*)(out + ((size_t)b * LOUT + t) * COUT + o0) = res;
}

// ---------------------------------------------------------------------------
// VQ: dist = ||z||^2 + ||e||^2 - 2 z.e, fused argmin over 4096 codes.
// Block = 8 waves, each wave owns 32 z-rows (A held in 64 VGPRs).
// The 4KB codebook tile (16 codes x 128 bf16) is shared by all 8 waves:
// double-buffered in LDS via CDNA5 async global->LDS copies (ASYNCcnt),
// consumed through ds_load_b128. 8 WMMAs per tile per wave.
// Running min/argmin kept in C-layout; shfl_xor butterfly within each
// 16-lane half (matches C/D lane split) finishes the reduction.
// ---------------------------------------------------------------------------
__global__ void __launch_bounds__(256)
k_vq_argmin(const __bf16* __restrict__ z, const float* __restrict__ zsq,
            const __bf16* __restrict__ cb, const float* __restrict__ esq,
            int* __restrict__ codes, float* __restrict__ mdist) {
  const int tid  = threadIdx.x;
  const int lane = tid & 31;
  const int wv   = tid >> 5;                       // wave in block, 0..7
  const int rowbase = (blockIdx.x * 8 + wv) * 32;  // 32 rows per wave
  if (rowbase >= NROWS) return;
  const int n   = lane & 15;
  const int hi8 = (lane >> 4) << 3;

  __shared__ __align__(16) __bf16 Bt[2][16 * DDIM];   // 2 x 4KB tiles

  // Persistent A fragments: 2 row-strips x 4 K-chunks = 64 VGPRs.
  v16bf a[2][4];
#pragma unroll
  for (int s = 0; s < 2; ++s) {
    const __bf16* ar = z + (size_t)(rowbase + s * 16 + n) * DDIM;
#pragma unroll
    for (int c = 0; c < 4; ++c) a[s][c] = load_frag(ar, c * 32, lane);
  }

  float zs[2][8];
#pragma unroll
  for (int s = 0; s < 2; ++s)
#pragma unroll
    for (int j = 0; j < 8; ++j) zs[s][j] = zsq[rowbase + s * 16 + hi8 + j];

  float minv[2][8];
  int   mini[2][8];
#pragma unroll
  for (int s = 0; s < 2; ++s)
#pragma unroll
    for (int j = 0; j < 8; ++j) { minv[s][j] = 3.4e38f; mini[s][j] = 0; }

  // Cooperative tile stage: 256 threads x 16B = 4KB tile.
  const int srow = tid >> 4, schunk = (tid & 15) << 3;   // row 0..15, elem chunk
  {
    const __bf16* src = cb + (size_t)srow * DDIM + schunk;
    async_copy_b128(src, &Bt[0][srow * DDIM + schunk]);
    wait_async0();
    __syncthreads();
  }

#pragma unroll 1
  for (int nt = 0; nt < KCODES / 16; ++nt) {
    const int cur = nt & 1;
    if (nt + 1 < KCODES / 16) {
      const __bf16* src = cb + (size_t)((nt + 1) * 16 + srow) * DDIM + schunk;
      async_copy_b128(src, &Bt[1 - cur][srow * DDIM + schunk]);
    }

    const __bf16* brow = &Bt[cur][n * DDIM];
    v16bf b0 = load_frag(brow,  0, lane);
    v16bf b1 = load_frag(brow, 32, lane);
    v16bf b2 = load_frag(brow, 64, lane);
    v16bf b3 = load_frag(brow, 96, lane);

    const int code = nt * 16 + n;
    const float es = esq[code];

#pragma unroll
    for (int s = 0; s < 2; ++s) {
      v8f acc;
#pragma unroll
      for (int j = 0; j < 8; ++j) acc[j] = 0.0f;
      acc = __builtin_amdgcn_wmma_f32_16x16x32_bf16(false, a[s][0], false, b0, (short)0, acc, false, false);
      acc = __builtin_amdgcn_wmma_f32_16x16x32_bf16(false, a[s][1], false, b1, (short)0, acc, false, false);
      acc = __builtin_amdgcn_wmma_f32_16x16x32_bf16(false, a[s][2], false, b2, (short)0, acc, false, false);
      acc = __builtin_amdgcn_wmma_f32_16x16x32_bf16(false, a[s][3], false, b3, (short)0, acc, false, false);
#pragma unroll
      for (int j = 0; j < 8; ++j) {
        float d = zs[s][j] + es - 2.0f * acc[j];
        if (d < minv[s][j]) { minv[s][j] = d; mini[s][j] = code; }  // '<' keeps first
      }
    }

    wait_async0();      // next tile landed in LDS
    __syncthreads();    // everyone done reading cur before it is restaged
  }

  // Butterfly min+argmin within each 16-lane half (xor masks < 16).
#pragma unroll
  for (int off = 1; off < 16; off <<= 1) {
#pragma unroll
    for (int s = 0; s < 2; ++s)
#pragma unroll
      for (int j = 0; j < 8; ++j) {
        float ov = __shfl_xor(minv[s][j], off, 32);
        int   oi = __shfl_xor(mini[s][j], off, 32);
        if (ov < minv[s][j] || (ov == minv[s][j] && oi < mini[s][j])) {
          minv[s][j] = ov; mini[s][j] = oi;
        }
      }
  }

  if (n == 0) {  // lanes 0 (rows +0..7) and 16 (rows +8..15)
#pragma unroll
    for (int s = 0; s < 2; ++s)
#pragma unroll
      for (int j = 0; j < 8; ++j) {
        int r = rowbase + s * 16 + hi8 + j;
        codes[r] = mini[s][j];
        mdist[r] = minv[s][j];
      }
  }
}

// ---------------------------------------------------------------------------
// Host launcher
// ---------------------------------------------------------------------------
extern "C" void kernel_launch(void* const* d_in, const int* in_sizes, int n_in,
                              void* d_out, int out_size, void* d_ws, size_t ws_size,
                              hipStream_t stream) {
  (void)in_sizes; (void)n_in; (void)out_size; (void)ws_size;

  const float* x   = (const float*)d_in[0];
  const float* w1  = (const float*)d_in[1];
  const float* b1  = (const float*)d_in[2];
  const float* g1  = (const float*)d_in[3];
  const float* be1 = (const float*)d_in[4];
  const float* m1  = (const float*)d_in[5];
  const float* v1  = (const float*)d_in[6];
  const float* w2  = (const float*)d_in[7];
  const float* b2  = (const float*)d_in[8];
  const float* g2  = (const float*)d_in[9];
  const float* be2 = (const float*)d_in[10];
  const float* m2  = (const float*)d_in[11];
  const float* v2  = (const float*)d_in[12];
  const float* w3  = (const float*)d_in[13];
  const float* b3  = (const float*)d_in[14];
  const float* cb  = (const float*)d_in[15];

  // Workspace pool (regions reused across pipeline stages):
  //   region A: xcl (64MB) -> later h2 (32MB)
  //   region B: h1 (32MB)  -> later z (8MB) + zsq
  char* ws = (char*)d_ws;
  const size_t OFF_A = 0;
  const size_t OFF_B = (size_t)BB * L_IN * C_IN0 * 2;            // 64 MB
  size_t off = OFF_B + (size_t)BB * L1D * C1D * 2;               // + 32 MB
  auto take = [&](size_t bytes) { size_t o = off; off = (off + bytes + 255) & ~(size_t)255; return o; };

  __bf16* xcl   = (__bf16*)(ws + OFF_A);
  __bf16* h1    = (__bf16*)(ws + OFF_B);
  __bf16* h2    = (__bf16*)(ws + OFF_A);                          // reuse A
  __bf16* zb    = (__bf16*)(ws + OFF_B);                          // reuse B
  float*  zsq   = (float*)(ws + OFF_B + (size_t)NROWS * DDIM * 2);

  __bf16* wk1   = (__bf16*)(ws + take((size_t)3 * C1D * C_IN0 * 2));
  float*  bias1 = (float*) (ws + take((size_t)C1D * 4));
  __bf16* wk2   = (__bf16*)(ws + take((size_t)3 * C2D * C1D * 2));
  float*  bias2 = (float*) (ws + take((size_t)C2D * 4));
  __bf16* wk3   = (__bf16*)(ws + take((size_t)3 * C3D * C2D * 2));
  float*  bias3 = (float*) (ws + take((size_t)C3D * 4));
  __bf16* cbb   = (__bf16*)(ws + take((size_t)KCODES * DDIM * 2));
  float*  esq   = (float*) (ws + take((size_t)KCODES * 4));

  int*   codes = (int*)d_out;
  float* md    = ((float*)d_out) + NROWS;

  // --- prep ---
  {
    size_t total = (size_t)BB * C_IN0 * L_IN;
    k_transpose_x<<<(unsigned)((total + 255) / 256), 256, 0, stream>>>(x, xcl);
  }
  k_prep_w<<<(C1D * C_IN0 * 3 + 255) / 256, 256, 0, stream>>>(
      w1, b1, g1, be1, m1, v1, wk1, bias1, C1D, C_IN0, 1);
  k_prep_w<<<(C2D * C1D * 3 + 255) / 256, 256, 0, stream>>>(
      w2, b2, g2, be2, m2, v2, wk2, bias2, C2D, C1D, 1);
  k_prep_w<<<(C3D * C2D * 3 + 255) / 256, 256, 0, stream>>>(
      w3, b3, b3, b3, b3, b3, wk3, bias3, C3D, C2D, 0);
  k_prep_cb<<<(KCODES + 127) / 128, 128, 0, stream>>>(cb, cbb, esq);

  // --- conv stack (implicit-GEMM WMMA) ---
  // tiles = B * (Lout/16) * (Cout/16); 8 waves per 256-thread block
  k_conv_wmma<C_IN0, C1D, L_IN, L1D, 2, true>
      <<<(BB * (L1D / 16) * (C1D / 16)) / 8, 256, 0, stream>>>(xcl, wk1, bias1, h1);
  k_conv_wmma<C1D, C2D, L1D, L2D, 2, true>
      <<<(BB * (L2D / 16) * (C2D / 16)) / 8, 256, 0, stream>>>(h1, wk2, bias2, h2);
  k_conv_wmma<C2D, C3D, L2D, L3D, 1, false>
      <<<(BB * (L3D / 16) * (C3D / 16)) / 8, 256, 0, stream>>>(h2, wk3, bias3, zb);

  // --- VQ (256 rows per block: 8 waves x 32 rows, LDS double-buffered B) ---
  k_row_sq<<<(NROWS + 255) / 256, 256, 0, stream>>>(zb, zsq);
  k_vq_argmin<<<NROWS / 256, 256, 0, stream>>>(zb, zsq, cbb, esq, codes, md);
}
